// mp_gnn_distributed_39943195852847
// MI455X (gfx1250) — compile-verified, matrix operand order checked by reading, not run on hardware
//
#include <hip/hip_runtime.h>

#define NN 100000
#define EE 800000
#define CC 64

typedef __attribute__((ext_vector_type(16))) __bf16         v16bf;
typedef __attribute__((ext_vector_type(16))) unsigned short v16us;
typedef __attribute__((ext_vector_type(8)))  float          v8f;

// ---- CDNA5 feature probes (device pass only) ------------------------------
#if defined(__gfx1250__)
#if __has_builtin(__builtin_amdgcn_global_load_async_to_lds_b128)
#define HAVE_ASYNC_LDS 1
#endif
#if __has_builtin(__builtin_amdgcn_tensor_load_to_lds) && \
    __has_builtin(__builtin_amdgcn_s_wait_tensorcnt)
#define HAVE_TDM 1
#endif
#endif

#if defined(HAVE_TDM)
typedef unsigned int u32x4 __attribute__((ext_vector_type(4)));
typedef int          i32x4 __attribute__((ext_vector_type(4)));
typedef int          i32x8 __attribute__((ext_vector_type(8)));

// 1-D TDM copy: nelem8 elements of 8 bytes, global -> LDS. D# per ISA ch.8.
__device__ __forceinline__ void tdm_load_1d(void* lds, const void* gsrc,
                                            unsigned nelem8) {
  unsigned long long ga = (unsigned long long)gsrc;
  unsigned lds_off = (unsigned)(unsigned long long)lds;
  u32x4 g0;
  g0[0] = 1u;                                     // count=1, user mode
  g0[1] = lds_off;                                // lds_addr
  g0[2] = (unsigned)ga;                           // global_addr[31:0]
  g0[3] = (unsigned)((ga >> 32) & 0x01FFFFFFull)  // global_addr[56:32]
          | (2u << 30);                           // type=2 ("image")
  i32x8 g1;
  g1[0] = (3 << 16);                              // wg_mask=0, data_size=8B
  g1[1] = (int)((nelem8 & 0xFFFFu) << 16);        // tensor_dim0[15:0]
  g1[2] = (int)((nelem8 >> 16) & 0xFFFFu) | (1 << 16); // dim0 hi | tensor_dim1=1
  g1[3] = (int)((nelem8 & 0xFFFFu) << 16);        // tile_dim0
  g1[4] = 0;                                      // tile_dim1/2 unused
  g1[5] = (int)nelem8;                            // tensor_dim0_stride lo
  g1[6] = 0;
  g1[7] = 0;
  i32x4 z4 = {0, 0, 0, 0};
  i32x8 z8 = {0, 0, 0, 0, 0, 0, 0, 0};
  __builtin_amdgcn_tensor_load_to_lds(g0, g1, z4, z4, z8, 0);
}
#endif

#if defined(HAVE_ASYNC_LDS)
typedef int gv4i __attribute__((vector_size(4 * sizeof(int))));
__device__ __forceinline__ void async_ld_b128(const void* g, void* l) {
  __builtin_amdgcn_global_load_async_to_lds_b128((gv4i*)g, (gv4i*)l, 0, 0);
}
__device__ __forceinline__ void wait_async0() {
#if __has_builtin(__builtin_amdgcn_s_wait_asynccnt)
  __builtin_amdgcn_s_wait_asynccnt(0);
#else
  asm volatile("s_wait_asynccnt 0x0" ::: "memory");
#endif
}
#endif

__device__ __forceinline__ unsigned short f2bf(float f) {
  unsigned u = __builtin_bit_cast(unsigned, f);
  u += 0x7FFFu + ((u >> 16) & 1u);          // round-to-nearest-even
  return (unsigned short)(u >> 16);
}

__device__ __forceinline__ float elu(float v) {
  return v > 0.f ? v : __expf(v) - 1.f;
}

// One 16x16x32 bf16 WMMA: A fragment from LDS (32B/lane), B fragment from
// packed weights (32B/lane, LDS when TDM-staged else global), f32 accumulate.
__device__ __forceinline__ v8f wmma_tile(const unsigned short* a,
                                         const unsigned short* b, v8f c) {
  v16bf av = __builtin_bit_cast(v16bf, *(const v16us*)a);
  v16bf bv = __builtin_bit_cast(v16bf, *(const v16us*)b);
  return __builtin_amdgcn_wmma_f32_16x16x32_bf16(false, av, false, bv,
                                                 (short)0, c, false, false);
}

__global__ void zero_kernel(float* __restrict__ p, int n) {
  int i = blockIdx.x * blockDim.x + threadIdx.x;
  if (i < n) p[i] = 0.f;
}

// Pack W[K][Nout] (f32 row-major) into bf16 WMMA B-fragments:
// [(kchunk*ntiles + ntile)][lane 0..31][16 contiguous K within lane half].
__global__ void pack_w_kernel(const float* __restrict__ W,
                              unsigned short* __restrict__ dst,
                              int K, int Nout, int ntiles, int total) {
  int idx = blockIdx.x * blockDim.x + threadIdx.x;
  if (idx >= total) return;
  int j = idx & 15;
  int L = (idx >> 4) & 31;
  int t = (idx >> 9) % ntiles;
  int c = idx / (512 * ntiles);
  int k = 32 * c + (L >> 4) * 16 + j;
  int n = 16 * t + (L & 15);
  float v = (k < K && n < Nout) ? W[k * Nout + n] : 0.f;
  dst[idx] = f2bf(v);
}

// ---------------- encoder: x[N,3] -> Lin(3,64)+ELU -> Lin(64,64) -> h ------
__global__ __launch_bounds__(128) void encoder_kernel(
    const float* __restrict__ x, const unsigned short* __restrict__ w1,
    const float* __restrict__ b1, const unsigned short* __restrict__ w2,
    const float* __restrict__ b2, float* __restrict__ h) {
  __shared__ __align__(32) unsigned short s_feat[16][32];
  __shared__ __align__(32) unsigned short s_hid[16][CC];
  int tid = threadIdx.x, wave = tid >> 5, lane = tid & 31;
  int row = lane & 15, khalf = lane >> 4;
  int n0b = blockIdx.x * 16;

  for (int i = tid; i < 16 * 32; i += 128) {
    int r = i >> 5, f = i & 31;
    s_feat[r][f] = f2bf(f < 3 ? x[(n0b + r) * 3 + f] : 0.f);
  }
  __syncthreads();

  int n0 = wave * 16;
  v8f acc = {};
  acc = wmma_tile(&s_feat[row][khalf * 16], &w1[(wave * 32 + lane) * 16], acc);
  float bias = b1[n0 + row];
  for (int r = 0; r < 8; ++r)
    s_hid[r + khalf * 8][n0 + row] = f2bf(elu(acc[r] + bias));
  __syncthreads();

  v8f acc2 = {};
  for (int c = 0; c < 2; ++c)
    acc2 = wmma_tile(&s_hid[row][c * 32 + khalf * 16],
                     &w2[((c * 4 + wave) * 32 + lane) * 16], acc2);
  float bias2 = b2[n0 + row];
  for (int r = 0; r < 8; ++r)
    h[(n0b + r + khalf * 8) * CC + n0 + row] = acc2[r] + bias2;
}

// ------- edge MLP + weighting + scatter-add, one block = 16 edges ----------
// w1 (7x4 fragments) and w2 (2x4 fragments) are CONTIGUOUS: w2 == w1 + 14336.
__global__ __launch_bounds__(128) void edge_kernel(
    const float* __restrict__ h, const int* __restrict__ ei,
    const float* __restrict__ ew, const float* __restrict__ pos,
    const unsigned short* __restrict__ w1, const float* __restrict__ b1,
    const unsigned short* __restrict__ w2, const float* __restrict__ b2,
    float* __restrict__ agg) {
  __shared__ int   s_src[16], s_dst[16];
  __shared__ float s_ew[16];
  __shared__ __align__(16) float s_hs[16][CC], s_hd[16][CC];
  __shared__ __align__(32) unsigned short s_feat[16][224];  // 195 pad to 7*32
  __shared__ __align__(32) unsigned short s_hid[16][CC];
#if defined(HAVE_TDM)
  __shared__ __align__(64) unsigned short s_w[18432];       // w1(14336)+w2(4096)
#endif
  int tid = threadIdx.x, wave = tid >> 5, lane = tid & 31;
  int row = lane & 15, khalf = lane >> 4;
  int e0 = blockIdx.x * 16;

#if defined(HAVE_TDM)
  if (wave == 0)                // one DMA for all weights, overlaps gather
    tdm_load_1d(&s_w[0], w1, 18432u / 4u);       // 4608 x 8B
#endif

  if (tid < 16) {
    s_src[tid] = ei[e0 + tid];
    s_dst[tid] = ei[EE + e0 + tid];
    s_ew[tid]  = ew[e0 + tid];
  }
  __syncthreads();

#if defined(HAVE_ASYNC_LDS)
  // gather h[src]/h[dst] straight into LDS, 16B per lane per op
  for (int i = tid; i < 256; i += 128) {
    int r = i >> 4, c = (i & 15) << 2;
    async_ld_b128(h + s_src[r] * CC + c, &s_hs[r][c]);
    async_ld_b128(h + s_dst[r] * CC + c, &s_hd[r][c]);
  }
  wait_async0();
  __syncthreads();
#else
  for (int i = tid; i < 16 * CC; i += 128) {
    int r = i >> 6, c = i & 63;
    s_hs[r][c] = h[s_src[r] * CC + c];
    s_hd[r][c] = h[s_dst[r] * CC + c];
  }
  __syncthreads();
#endif

  for (int i = tid; i < 16 * 224; i += 128) {      // concat feature build
    int r = i / 224, f = i - r * 224;
    float v;
    if (f < 64)       v = s_hs[r][f];
    else if (f < 128) v = s_hd[r][f - 64];
    else if (f < 192) v = s_hs[r][f - 128] - s_hd[r][f - 128];
    else if (f < 195) v = pos[s_src[r] * 3 + (f - 192)] -
                          pos[s_dst[r] * 3 + (f - 192)];
    else              v = 0.f;
    s_feat[r][f] = f2bf(v);
  }
#if defined(HAVE_TDM)
  __builtin_amdgcn_s_wait_tensorcnt(0);            // weights landed in LDS
#endif
  __syncthreads();

#if defined(HAVE_TDM)
  const unsigned short* w1p = &s_w[0];
  const unsigned short* w2p = &s_w[14336];
#else
  const unsigned short* w1p = w1;
  const unsigned short* w2p = w2;
#endif

  int n0 = wave * 16;
  v8f acc = {};
  for (int c = 0; c < 7; ++c)                      // 195->64
    acc = wmma_tile(&s_feat[row][c * 32 + khalf * 16],
                    &w1p[((c * 4 + wave) * 32 + lane) * 16], acc);
  float bias = b1[n0 + row];
  for (int r = 0; r < 8; ++r)
    s_hid[r + khalf * 8][n0 + row] = f2bf(elu(acc[r] + bias));
  __syncthreads();

  v8f acc2 = {};
  for (int c = 0; c < 2; ++c)                      // 64->64
    acc2 = wmma_tile(&s_hid[row][c * 32 + khalf * 16],
                     &w2p[((c * 4 + wave) * 32 + lane) * 16], acc2);
  float bias2 = b2[n0 + row];
  for (int r = 0; r < 8; ++r) {
    int m = r + khalf * 8;
    float v = (acc2[r] + bias2) * s_ew[m];
    atomicAdd(&agg[s_dst[m] * CC + n0 + row], v);  // segment_sum
  }
}

// ------------- node MLP + residual, one block = 16 nodes -------------------
__global__ __launch_bounds__(128) void node_kernel(
    const float* __restrict__ h_in, const float* __restrict__ agg,
    const unsigned short* __restrict__ w1, const float* __restrict__ b1,
    const unsigned short* __restrict__ w2, const float* __restrict__ b2,
    float* __restrict__ h_out) {
  __shared__ __align__(32) unsigned short s_feat[16][128];
  __shared__ __align__(32) unsigned short s_hid[16][CC];
  int tid = threadIdx.x, wave = tid >> 5, lane = tid & 31;
  int row = lane & 15, khalf = lane >> 4;
  int n0b = blockIdx.x * 16;

  for (int i = tid; i < 16 * 128; i += 128) {
    int r = i >> 7, f = i & 127;
    float v = (f < CC) ? h_in[(n0b + r) * CC + f]
                       : agg[(n0b + r) * CC + (f - CC)];
    s_feat[r][f] = f2bf(v);
  }
  __syncthreads();

  int n0 = wave * 16;
  v8f acc = {};
  for (int c = 0; c < 4; ++c)                      // 128->64
    acc = wmma_tile(&s_feat[row][c * 32 + khalf * 16],
                    &w1[((c * 4 + wave) * 32 + lane) * 16], acc);
  float bias = b1[n0 + row];
  for (int r = 0; r < 8; ++r)
    s_hid[r + khalf * 8][n0 + row] = f2bf(elu(acc[r] + bias));
  __syncthreads();

  v8f acc2 = {};
  for (int c = 0; c < 2; ++c)                      // 64->64
    acc2 = wmma_tile(&s_hid[row][c * 32 + khalf * 16],
                     &w2[((c * 4 + wave) * 32 + lane) * 16], acc2);
  float bias2 = b2[n0 + row];
  for (int r = 0; r < 8; ++r) {
    int m = r + khalf * 8;
    int gi = (n0b + m) * CC + n0 + row;
    h_out[gi] = h_in[gi] + acc2[r] + bias2;        // residual
  }
}

// ------------- decoder: Lin(64,64)+ELU -> Lin(64,3) ------------------------
__global__ __launch_bounds__(128) void decoder_kernel(
    const float* __restrict__ h, const unsigned short* __restrict__ w1,
    const float* __restrict__ b1, const unsigned short* __restrict__ w2,
    const float* __restrict__ b2, float* __restrict__ out) {
  __shared__ __align__(32) unsigned short s_feat[16][CC];
  __shared__ __align__(32) unsigned short s_hid[16][CC];
  int tid = threadIdx.x, wave = tid >> 5, lane = tid & 31;
  int row = lane & 15, khalf = lane >> 4;
  int n0b = blockIdx.x * 16;

  for (int i = tid; i < 16 * CC; i += 128) {
    int r = i >> 6, f = i & 63;
    s_feat[r][f] = f2bf(h[(n0b + r) * CC + f]);
  }
  __syncthreads();

  int n0 = wave * 16;
  v8f acc = {};
  for (int c = 0; c < 2; ++c)
    acc = wmma_tile(&s_feat[row][c * 32 + khalf * 16],
                    &w1[((c * 4 + wave) * 32 + lane) * 16], acc);
  float bias = b1[n0 + row];
  for (int r = 0; r < 8; ++r)
    s_hid[r + khalf * 8][n0 + row] = f2bf(elu(acc[r] + bias));
  __syncthreads();

  if (wave == 0) {                                 // 64 -> 3 (padded to 16)
    v8f acc2 = {};
    for (int c = 0; c < 2; ++c)
      acc2 = wmma_tile(&s_hid[row][c * 32 + khalf * 16],
                       &w2[(c * 32 + lane) * 16], acc2);
    if (row < 3) {
      float bias2 = b2[row];
      for (int r = 0; r < 8; ++r)
        out[(n0b + r + khalf * 8) * 3 + row] = acc2[r] + bias2;
    }
  }
}

extern "C" void kernel_launch(void* const* d_in, const int* in_sizes, int n_in,
                              void* d_out, int out_size, void* d_ws, size_t ws_size,
                              hipStream_t stream) {
  (void)in_sizes; (void)n_in; (void)out_size; (void)ws_size;
  const float* x   = (const float*)d_in[0];
  const int*   ei  = (const int*)d_in[1];
  const float* ew  = (const float*)d_in[2];
  const float* pos = (const float*)d_in[3];
  // params flattened jax tree_leaves order (sorted dict keys): dec, enc, mp
  const float* decW1 = (const float*)d_in[4];
  const float* decB1 = (const float*)d_in[5];
  const float* decW2 = (const float*)d_in[6];
  const float* decB2 = (const float*)d_in[7];
  const float* encW1 = (const float*)d_in[8];
  const float* encB1 = (const float*)d_in[9];
  const float* encW2 = (const float*)d_in[10];
  const float* encB2 = (const float*)d_in[11];
  const float *eW1[2], *eB1[2], *eW2[2], *eB2[2];
  const float *nW1[2], *nB1[2], *nW2[2], *nB2[2];
  for (int l = 0; l < 2; ++l) {
    int b = 12 + l * 8;
    eW1[l] = (const float*)d_in[b + 0]; eB1[l] = (const float*)d_in[b + 1];
    eW2[l] = (const float*)d_in[b + 2]; eB2[l] = (const float*)d_in[b + 3];
    nW1[l] = (const float*)d_in[b + 4]; nB1[l] = (const float*)d_in[b + 5];
    nW2[l] = (const float*)d_in[b + 6]; nB2[l] = (const float*)d_in[b + 7];
  }

  // workspace: hA | hB | agg | packed bf16 weights
  float* hA  = (float*)d_ws;
  float* hB  = hA + (size_t)NN * CC;
  float* agg = hB + (size_t)NN * CC;
  unsigned short* wp = (unsigned short*)(agg + (size_t)NN * CC);
  unsigned short* p_enc1 = wp;                     // 1 chunk x 4 tiles
  unsigned short* p_enc2 = p_enc1 + 1 * 4 * 512;   // 2 x 4
  unsigned short* p_e1[2], *p_e2[2], *p_n1[2], *p_n2[2];
  unsigned short* cur = p_enc2 + 2 * 4 * 512;
  for (int l = 0; l < 2; ++l) {
    p_e1[l] = cur; cur += 7 * 4 * 512;  // edge w1; w2 contiguous right after:
    p_e2[l] = cur; cur += 2 * 4 * 512;  // (TDM copies w1..w2 as one tile)
    p_n1[l] = cur; cur += 4 * 4 * 512;
    p_n2[l] = cur; cur += 2 * 4 * 512;
  }
  unsigned short* p_d1 = cur; cur += 2 * 4 * 512;
  unsigned short* p_d2 = cur; cur += 2 * 1 * 512;  // 64 -> 3 (1 tile)

  auto pack = [&](const float* W, unsigned short* dst, int K, int Nout,
                  int ntiles, int kchunks) {
    int total = kchunks * ntiles * 512;
    pack_w_kernel<<<(total + 255) / 256, 256, 0, stream>>>(W, dst, K, Nout,
                                                           ntiles, total);
  };
  pack(encW1, p_enc1, 3, 64, 4, 1);
  pack(encW2, p_enc2, 64, 64, 4, 2);
  for (int l = 0; l < 2; ++l) {
    pack(eW1[l], p_e1[l], 195, 64, 4, 7);
    pack(eW2[l], p_e2[l], 64, 64, 4, 2);
    pack(nW1[l], p_n1[l], 128, 64, 4, 4);
    pack(nW2[l], p_n2[l], 64, 64, 4, 2);
  }
  pack(decW1, p_d1, 64, 64, 4, 2);
  pack(decW2, p_d2, 64, 3, 1, 2);

  encoder_kernel<<<NN / 16, 128, 0, stream>>>(x, p_enc1, encB1, p_enc2, encB2, hA);

  float* hcur = hA; float* hnext = hB;
  for (int l = 0; l < 2; ++l) {
    zero_kernel<<<(NN * CC + 255) / 256, 256, 0, stream>>>(agg, NN * CC);
    edge_kernel<<<EE / 16, 128, 0, stream>>>(hcur, ei, ew, pos, p_e1[l], eB1[l],
                                             p_e2[l], eB2[l], agg);
    node_kernel<<<NN / 16, 128, 0, stream>>>(hcur, agg, p_n1[l], nB1[l],
                                             p_n2[l], nB2[l], hnext);
    float* t = hcur; hcur = hnext; hnext = t;
  }
  decoder_kernel<<<NN / 16, 128, 0, stream>>>(hcur, p_d1, decB1, p_d2, decB2,
                                              (float*)d_out);
}
